// GradCacheColbertPairwiseNegativeCELoss_9139690406184
// MI455X (gfx1250) — compile-verified, matrix operand order checked by reading, not run on hardware
//
#include <hip/hip_runtime.h>
#include <hip/hip_bf16.h>

// Problem sizes (fixed by reference)
#define B_ 128
#define N_ 32
#define S_ 1024
#define D_ 128
#define CH_ 128   // doc tokens staged per LDS chunk (32 KB per buffer)

typedef __attribute__((ext_vector_type(16))) __bf16 bf16x16;
typedef __attribute__((ext_vector_type(8)))  __bf16 bf16x8;
typedef __attribute__((ext_vector_type(8)))  float  f32x8;

// ---------------------------------------------------------------------------
// Kernel 0: f32 -> bf16 conversion (8 elements / thread, vectorized)
// ---------------------------------------------------------------------------
__global__ void cvt_f32_to_bf16(const float* __restrict__ in,
                                __bf16* __restrict__ out, int n) {
    int i = (blockIdx.x * blockDim.x + threadIdx.x) * 8;
    if (i + 8 <= n) {
        const float4* p = (const float4*)(in + i);
        float4 a = p[0];
        float4 b = p[1];
        bf16x8 o;
        o[0] = (__bf16)a.x; o[1] = (__bf16)a.y; o[2] = (__bf16)a.z; o[3] = (__bf16)a.w;
        o[4] = (__bf16)b.x; o[5] = (__bf16)b.y; o[6] = (__bf16)b.z; o[7] = (__bf16)b.w;
        *(bf16x8*)(out + i) = o;
    }
}

__device__ __forceinline__ float softplus_f(float x) {
    return fmaxf(x, 0.0f) + log1pf(expf(-fabsf(x)));
}

// ---------------------------------------------------------------------------
// Async global->LDS stage of one 32 KB chunk (CH_ x D_ bf16) with 256 threads.
// Each thread moves 8 x 16 B via GLOBAL_LOAD_ASYNC_TO_LDS_B128 (ASYNCcnt).
// lbase = low 32 bits of the generic shared-memory pointer = LDS byte offset.
// ---------------------------------------------------------------------------
__device__ __forceinline__ void stage_async(const __bf16* __restrict__ g,
                                            unsigned lbase, int tid) {
#pragma unroll
    for (int i = 0; i < 8; ++i) {
        unsigned off = (unsigned)((i * 256 + tid) * 16);
        unsigned la = lbase + off;
        unsigned long long ga = (unsigned long long)g + off;
        asm volatile("global_load_async_to_lds_b128 %0, %1, off"
                     :: "v"(la), "v"(ga) : "memory");
    }
}

__device__ __forceinline__ void wait_async0() {
    asm volatile("s_wait_asynccnt 0x0" ::: "memory");
}

// ---------------------------------------------------------------------------
// Kernel 1: in-batch cross scores.
//   out[b*B+c] = sum_n max_s dot(q[b,n,:], dk[c,s,:])
// Grid (c, bgroup): block owns doc c and 8 queries b = bgroup*8 + wave.
// Block = 256 threads = 8 wave32s; wave w computes the FULL [32 x 1024]
// MaxSim for its b: it holds A fragments for both 16-row M-tiles
// (2 mtiles x 4 k-steps = 64 VGPRs), so each B fragment read from LDS
// feeds 2 WMMAs. d[c] is streamed once per block through a double-buffered
// 64 KB LDS window filled by async DMA overlapped with compute.
// ---------------------------------------------------------------------------
__global__ __launch_bounds__(256)
void maxsim_cross(const __bf16* __restrict__ q,    // [B][N][D]
                  const __bf16* __restrict__ dk,   // [B][S][D]
                  float* __restrict__ out) {       // [B][B]
    const int c    = blockIdx.x;
    const int bg   = blockIdx.y;            // 0..15
    const int tid  = threadIdx.x;
    const int lane = tid & 31;
    const int wave = tid >> 5;              // 0..7
    const int b    = bg * 8 + wave;

    __shared__ __align__(16) __bf16 lds[2][CH_ * D_];  // 2 x 32 KB

    // ---- Preload A (query) fragments for both M-tiles.
    // Row = mt*16 + lane%16; lanes 0-15 take K [k0,k0+16), lanes 16-31 [k0+16,k0+32).
    const int khalf = (lane >> 4) << 4;  // 0 or 16
    bf16x16 afrag[2][4];
#pragma unroll
    for (int mt = 0; mt < 2; ++mt) {
        const __bf16* qp =
            q + ((size_t)b * N_ + (size_t)(mt * 16 + (lane & 15))) * D_ + khalf;
#pragma unroll
        for (int ks = 0; ks < 4; ++ks)
            afrag[mt][ks] = *(const bf16x16*)(qp + ks * 32);
    }

    float rowmax[2][8];
#pragma unroll
    for (int mt = 0; mt < 2; ++mt)
#pragma unroll
        for (int i = 0; i < 8; ++i) rowmax[mt][i] = -3.0e38f;

    const __bf16* dbase = dk + (size_t)c * S_ * D_;
    const unsigned lb[2] = {(unsigned)(size_t)&lds[0][0],
                            (unsigned)(size_t)&lds[1][0]};

    // ---- Prologue: stage chunk 0
    stage_async(dbase, lb[0], tid);
    wait_async0();
    __syncthreads();

    for (int ch = 0; ch < S_ / CH_; ++ch) {
        const int buf = ch & 1;
        if (ch + 1 < S_ / CH_)
            stage_async(dbase + (size_t)(ch + 1) * CH_ * D_, lb[buf ^ 1], tid);

        // ---- Compute on lds[buf]: 8 column tiles of 16 doc tokens.
#pragma unroll
        for (int t = 0; t < CH_ / 16; ++t) {
            const __bf16* lp = &lds[buf][(size_t)(t * 16 + (lane & 15)) * D_ + khalf];
            f32x8 acc0 = {0.f, 0.f, 0.f, 0.f, 0.f, 0.f, 0.f, 0.f};
            f32x8 acc1 = {0.f, 0.f, 0.f, 0.f, 0.f, 0.f, 0.f, 0.f};
#pragma unroll
            for (int ks = 0; ks < 4; ++ks) {
                bf16x16 bfrag = *(const bf16x16*)(lp + ks * 32);
                acc0 = __builtin_amdgcn_wmma_f32_16x16x32_bf16(
                    false, afrag[0][ks], false, bfrag, (short)0, acc0, false, false);
                acc1 = __builtin_amdgcn_wmma_f32_16x16x32_bf16(
                    false, afrag[1][ks], false, bfrag, (short)0, acc1, false, false);
            }
#pragma unroll
            for (int i = 0; i < 8; ++i) {
                rowmax[0][i] = fmaxf(rowmax[0][i], acc0[i]);
                rowmax[1][i] = fmaxf(rowmax[1][i], acc1[i]);
            }
        }

        // Wait for next chunk's DMA, then rendezvous before switching buffers.
        wait_async0();
        __syncthreads();
    }

    // ---- Reduce: max over the 16 columns held across each 16-lane half,
    // then sum over rows. C layout: VGPR i, lanes 0-15 -> row i, lanes 16-31 -> row i+8.
    float total = 0.0f;
#pragma unroll
    for (int mt = 0; mt < 2; ++mt) {
#pragma unroll
        for (int i = 0; i < 8; ++i) {
            float v = rowmax[mt][i];
            v = fmaxf(v, __shfl_xor(v, 1, 32));
            v = fmaxf(v, __shfl_xor(v, 2, 32));
            v = fmaxf(v, __shfl_xor(v, 4, 32));
            v = fmaxf(v, __shfl_xor(v, 8, 32));
            total += v;  // half 0: rows mt*16+0..7, half 1: rows mt*16+8..15
        }
    }
    total += __shfl_xor(total, 16, 32);  // combine the two halves
    if (lane == 0) out[(size_t)b * B_ + c] = total;
}

// ---------------------------------------------------------------------------
// Kernel 2: pairwise negatives, out[b] = sum_n max_s dot(q[b,n],nd[b,s]).
// Only B pairs (1/128 of the cross work) -> simple direct-from-L2 version.
// Block = 256 threads = 8 waves = 2 mtiles x 4 column groups.
// ---------------------------------------------------------------------------
__global__ __launch_bounds__(256)
void maxsim_pair(const __bf16* __restrict__ q,
                 const __bf16* __restrict__ dk,
                 float* __restrict__ out) {
    const int b     = blockIdx.x;
    const int lane  = threadIdx.x & 31;
    const int wave  = threadIdx.x >> 5;
    const int mtile = wave & 1;
    const int cgrp  = wave >> 1;

    __shared__ float red[32][4];

    const int khalf = (lane >> 4) << 4;
    const __bf16* qp =
        q + ((size_t)b * N_ + (size_t)(mtile * 16 + (lane & 15))) * D_ + khalf;
    bf16x16 afrag[4];
#pragma unroll
    for (int ks = 0; ks < 4; ++ks)
        afrag[ks] = *(const bf16x16*)(qp + ks * 32);

    const __bf16* dbase =
        dk + (size_t)b * S_ * D_ + (size_t)(lane & 15) * D_ + khalf;

    float rowmax[8];
#pragma unroll
    for (int i = 0; i < 8; ++i) rowmax[i] = -3.0e38f;

    for (int t = 0; t < 16; ++t) {
        const __bf16* dptr = dbase + (size_t)(cgrp * 256 + t * 16) * D_;
        f32x8 acc = {0.f, 0.f, 0.f, 0.f, 0.f, 0.f, 0.f, 0.f};
#pragma unroll
        for (int ks = 0; ks < 4; ++ks) {
            bf16x16 bfrag = *(const bf16x16*)(dptr + ks * 32);
            acc = __builtin_amdgcn_wmma_f32_16x16x32_bf16(
                false, afrag[ks], false, bfrag, (short)0, acc, false, false);
        }
#pragma unroll
        for (int i = 0; i < 8; ++i) rowmax[i] = fmaxf(rowmax[i], acc[i]);
    }

#pragma unroll
    for (int i = 0; i < 8; ++i) {
        float v = rowmax[i];
        v = fmaxf(v, __shfl_xor(v, 1, 32));
        v = fmaxf(v, __shfl_xor(v, 2, 32));
        v = fmaxf(v, __shfl_xor(v, 4, 32));
        v = fmaxf(v, __shfl_xor(v, 8, 32));
        rowmax[i] = v;
    }
    if ((lane & 15) == 0) {
        const int rbase = mtile * 16 + ((lane >> 4) ? 8 : 0);
#pragma unroll
        for (int i = 0; i < 8; ++i) red[rbase + i][cgrp] = rowmax[i];
    }
    __syncthreads();

    if (threadIdx.x < 32) {
        float m = fmaxf(fmaxf(red[threadIdx.x][0], red[threadIdx.x][1]),
                        fmaxf(red[threadIdx.x][2], red[threadIdx.x][3]));
        m += __shfl_xor(m, 1, 32);
        m += __shfl_xor(m, 2, 32);
        m += __shfl_xor(m, 4, 32);
        m += __shfl_xor(m, 8, 32);
        m += __shfl_xor(m, 16, 32);
        if (threadIdx.x == 0) out[b] = m;
    }
}

// ---------------------------------------------------------------------------
// Kernel 3: final losses. scores[b,b] doubles as pos_scores[b].
// ---------------------------------------------------------------------------
__global__ void finalize_loss(const float* __restrict__ scores,  // [B][B]
                              const float* __restrict__ negp,    // [B]
                              float* __restrict__ out) {
    __shared__ float sh[B_];
    const int b = threadIdx.x;
    const float pos = scores[(size_t)b * B_ + b];

    float mx = -3.0e38f;
    for (int cc = 0; cc < B_; ++cc) {
        float v = scores[(size_t)b * B_ + cc];
        if (cc == b) v -= 1.0e6f;  // NEG_INF_DIAG mask
        mx = fmaxf(mx, v);
    }
    const float l1 = softplus_f(negp[b] - pos);
    const float l2 = softplus_f(mx - pos);
    sh[b] = l1 + l2;
    __syncthreads();
    for (int off = 64; off > 0; off >>= 1) {
        if (b < off) sh[b] += sh[b + off];
        __syncthreads();
    }
    if (b == 0) out[0] = sh[0] / (2.0f * (float)B_);
}

// ---------------------------------------------------------------------------
// Launch
// ---------------------------------------------------------------------------
extern "C" void kernel_launch(void* const* d_in, const int* in_sizes, int n_in,
                              void* d_out, int out_size, void* d_ws, size_t ws_size,
                              hipStream_t stream) {
    const float* q_f  = (const float*)d_in[0];  // [B,N,D]
    const float* d_f  = (const float*)d_in[1];  // [B,S,D]
    const float* nd_f = (const float*)d_in[2];  // [B,S,D]

    const size_t nQ = (size_t)B_ * N_ * D_;  // 524288
    const size_t nD = (size_t)B_ * S_ * D_;  // 16777216

    char* ws = (char*)d_ws;
    size_t off = 0;
    __bf16* q_b  = (__bf16*)(ws + off); off += (nQ * 2 + 255) & ~(size_t)255;
    __bf16* d_b  = (__bf16*)(ws + off); off += (nD * 2 + 255) & ~(size_t)255;
    __bf16* nd_b = (__bf16*)(ws + off); off += (nD * 2 + 255) & ~(size_t)255;
    float*  scores = (float*)(ws + off); off += ((size_t)B_ * B_ * 4 + 255) & ~(size_t)255;
    float*  negp   = (float*)(ws + off); off += ((size_t)B_ * 4 + 255) & ~(size_t)255;

    // 1) convert inputs to bf16
    cvt_f32_to_bf16<<<(int)(nQ / 8 / 256), 256, 0, stream>>>(q_f, q_b, (int)nQ);
    cvt_f32_to_bf16<<<(int)(nD / 8 / 256), 256, 0, stream>>>(d_f, d_b, (int)nD);
    cvt_f32_to_bf16<<<(int)(nD / 8 / 256), 256, 0, stream>>>(nd_f, nd_b, (int)nD);

    // 2) full B x B in-batch MaxSim matrix (diag == pos_scores)
    maxsim_cross<<<dim3(B_, B_ / 8), 256, 0, stream>>>(q_b, d_b, scores);

    // 3) pairwise negatives
    maxsim_pair<<<dim3(B_), 256, 0, stream>>>(q_b, nd_b, negp);

    // 4) losses -> scalar
    finalize_loss<<<1, B_, 0, stream>>>(scores, negp, (float*)d_out);
}
